// Attention4D_21689584845120
// MI455X (gfx1250) — matrix operand order repaired and checked by reading, not compiled, not run on hardware
//
#include <hip/hip_runtime.h>
#include <hip/hip_fp16.h>

// ---------------------------------------------------------------------------
// Attention4D (EfficientViT-style) for MI455X / gfx1250, wave32 + WMMA + TDM.
//   B=64, DIM=384, RES=14 (N=196), HEADS=8, KEY_DIM=32, D=128, DH=1024, QK=256
// v3: as v2 (f16 K-contiguous packed operands, LDS-staged shared A-tiles,
// fused epilogues, global_prefetch_b8) plus:
//   * attn_v A-tile staged via the Tensor Data Mover (tensor_load_to_lds +
//     s_wait_tensorcnt), one DMA per workgroup instead of 28 vector copies.
//   * prefetch locality raised to near-cache (WGP) scope.
//
// Pipeline:
//   0a) pack_weights: wq|wk|wv -> w16[1536][384], wp -> wp16[384][1024] (f16)
//   0b) pack_x      : x[b,C,n] -> xt16[(b,n)][384] (f16, transposed)
//   1)  qkv_gemm    : WMMA GEMM 1536x384x12544 + bias + BN -> q,k,v (f32)
//                     (v additionally stored f16 K-major: v16[ch][224])
//   2)  dwconv_bn   : 3x3 depthwise conv on v + BN -> ht16[(b,n)][ch] (f16)
//   3)  attn_scores : QK^T*scale + pos-bias -> th1 -> softmax -> th2
//                     -> attn f16, row stride 224, pad zero-filled
//   4)  attn_v      : WMMA attn(16x224) @ v(224x16); out = relu(out+v_local)
//                     stored back into ht16 (proj's B layout)
//   5)  proj_gemm   : WMMA GEMM 384x1024x12544 + bias + BN -> d_out
// Workspace ~180 MB, fits the 192 MB L2.
// ---------------------------------------------------------------------------

typedef __attribute__((ext_vector_type(16))) _Float16 v16h;
typedef __attribute__((ext_vector_type(8)))  _Float16 v8h;
typedef __attribute__((ext_vector_type(8)))  float    v8f;
typedef __attribute__((ext_vector_type(4)))  unsigned int u32x4;
typedef __attribute__((ext_vector_type(8)))  int      i32x8;
typedef __attribute__((ext_vector_type(4)))  int      i32x4;

#define B_     64
#define DIMC   384
#define RES_   14
#define NTOK   196
#define HEADS_ 8
#define KD     32
#define DV     128
#define DH_    1024
#define QKC    256
#define AMST   224          // padded attn/v16 row stride (16B-aligned, K-pad)
#define EPSB   1e-5f
#define SCALE_ 0.17677669529663687f   // 32^-0.5

// ---------------------------------------------------------------------------
// TDM: load a 2-D tile (16-bit elements) from global into LDS.
// D# per CDNA5 ISA §8.3/8.4: group0 = {flags, lds_addr, global_addr, type=2},
// group1 = {data_size=2B, tensor_dim0/1, tile_dim0/1, tensor_dim0_stride}.
// Issued by one wave; completion via TENSORcnt.
// ---------------------------------------------------------------------------
__device__ __forceinline__ void tdm_load_tile_b16(
    void* lds_dst, const void* gsrc,
    unsigned tile_x, unsigned tile_y, unsigned row_stride_elems)
{
#if defined(__HIP_DEVICE_COMPILE__)
    const unsigned long long ga = (unsigned long long)(uintptr_t)gsrc;
    const unsigned la = (unsigned)(uintptr_t)lds_dst;   // low 32 bits = LDS offset

    u32x4 g0;
    g0[0] = 1u;                                   // count=1, user D#, no gather
    g0[1] = la;                                   // lds_addr          [63:32]
    g0[2] = (unsigned)ga;                         // global_addr lo    [95:64]
    g0[3] = (unsigned)(ga >> 32) | (2u << 30);    // global_addr hi, type=2

    i32x8 g1;
    g1[0] = (int)(1u << 16);                      // wg_mask=0, data_size=1 (2B)
    g1[1] = (int)((row_stride_elems & 0xFFFFu) << 16);        // tensor_dim0 lo
    g1[2] = (int)(((row_stride_elems >> 16) & 0xFFFFu)        // tensor_dim0 hi
                  | ((tile_y & 0xFFFFu) << 16));              // tensor_dim1 lo
    g1[3] = (int)((tile_x & 0xFFFFu) << 16);                  // tile_dim0
    g1[4] = (int)(tile_y & 0xFFFFu);                          // tile_dim1
    g1[5] = (int)row_stride_elems;                            // dim0_stride lo32
    g1[6] = 0;
    g1[7] = 0;

    i32x4 gz = {0, 0, 0, 0};                      // groups 2/3: unused (<=2D)
#if __clang_major__ >= 23
    i32x8 gz8 = {0, 0, 0, 0, 0, 0, 0, 0};
    __builtin_amdgcn_tensor_load_to_lds(g0, g1, gz, gz, gz8, 0);
#else
    __builtin_amdgcn_tensor_load_to_lds(g0, g1, gz, gz, 0);
#endif
    __builtin_amdgcn_s_wait_tensorcnt(0);
#else
    (void)lds_dst; (void)gsrc; (void)tile_x; (void)tile_y; (void)row_stride_elems;
#endif
}

// f16 WMMA fragment layouts (ISA 7.12.2):
//  A 16x32: lane half 'off' = (lane>=16 ? 8 : 0); elements 0-7 hold K=off+0..7,
//           elements 8-15 hold K=16+off+0..7  -> two aligned v8h loads.
//  B 32x16: same K order as A (klow = lane>=16 ? 8 : 0; K=klow+0..7, 16+klow+0..7)
//           -> two aligned v8h loads at +klow and +klow+16.
__device__ __forceinline__ v16h load_a_frag(const _Float16* rowp, int k0, int lane) {
    const int off = (lane & 16) ? 8 : 0;
    const v8h a0 = *(const v8h*)(rowp + k0 + off);
    const v8h a1 = *(const v8h*)(rowp + k0 + off + 16);
    v16h a;
#pragma unroll
    for (int e = 0; e < 8; ++e) { a[e] = a0[e]; a[e + 8] = a1[e]; }
    return a;
}
__device__ __forceinline__ v16h load_b_frag(const _Float16* colp, int k0, int lane) {
    const int klow = (lane & 16) ? 8 : 0;
    const v8h b0 = *(const v8h*)(colp + k0 + klow);
    const v8h b1 = *(const v8h*)(colp + k0 + klow + 16);
    v16h b;
#pragma unroll
    for (int e = 0; e < 8; ++e) { b[e] = b0[e]; b[e + 8] = b1[e]; }
    return b;
}

// ---------------------------------------------------------------------------
// Kernel 0a: pack weights to f16 (w16 = [wq;wk;wv] row-major, wp16 = wp)
// ---------------------------------------------------------------------------
__global__ __launch_bounds__(256)
void pack_weights(const float* __restrict__ wq, const float* __restrict__ wk,
                  const float* __restrict__ wv, const float* __restrict__ wp,
                  _Float16* __restrict__ w16, _Float16* __restrict__ wp16)
{
    const int idx = blockIdx.x * 256 + threadIdx.x;          // 983040 total
    if (idx < 1536 * DIMC) {
        const int row = idx / DIMC, c = idx % DIMC;
        float v;
        if (row < QKC)            v = wq[row * DIMC + c];
        else if (row < 2 * QKC)   v = wk[(row - QKC) * DIMC + c];
        else                      v = wv[(row - 2 * QKC) * DIMC + c];
        w16[idx] = (_Float16)v;
    } else {
        const int j = idx - 1536 * DIMC;                     // 384*1024
        wp16[j] = (_Float16)wp[j];
    }
}

// ---------------------------------------------------------------------------
// Kernel 0b: transpose+convert x -> xt16[(b*196+n)][384]
// ---------------------------------------------------------------------------
__global__ __launch_bounds__(256)
void pack_x(const float* __restrict__ x, _Float16* __restrict__ xt16)
{
    const size_t idx = (size_t)blockIdx.x * 256 + threadIdx.x;   // B*N*DIMC
    const int c = (int)(idx % DIMC);
    const int j = (int)(idx / DIMC);
    const int b = j / NTOK, n = j % NTOK;
    xt16[idx] = (_Float16)x[((size_t)b * DIMC + c) * NTOK + n];
}

// ---------------------------------------------------------------------------
// Kernel 1: fused QKV WMMA GEMM + bias + BN.  Block: 8 waves share one
// 16-row A-tile (staged in LDS); each wave owns a 16-col B slice.
// grid = 96 (o-tiles) * 98 (col-tile-octets) = 9408
// ---------------------------------------------------------------------------
__global__ __launch_bounds__(256)
void qkv_gemm(const _Float16* __restrict__ w16, const _Float16* __restrict__ xt16,
              const float* __restrict__ bq, const float* __restrict__ bnq,
              const float* __restrict__ bk, const float* __restrict__ bnk,
              const float* __restrict__ bv, const float* __restrict__ bnv,
              float* __restrict__ qws, float* __restrict__ kws,
              float* __restrict__ vws, _Float16* __restrict__ v16)
{
    __shared__ __align__(16) _Float16 As[16 * DIMC];             // 12 KB
    const int lane = threadIdx.x & 31;
    const int wave = threadIdx.x >> 5;
    const int ot   = blockIdx.x % 96;
    const int jt   = (blockIdx.x / 96) * 8 + wave;
    const int col  = jt * 16 + (lane & 15);                      // 0..12543
    const int b    = col / NTOK;
    const int n    = col % NTOK;
    const int orow0 = ot * 16;

    {   // cooperative A-tile copy: 16 rows x 384 halves, contiguous
        const uint4* src = (const uint4*)(w16 + (size_t)orow0 * DIMC);
        uint4* dst = (uint4*)As;
        for (int i = threadIdx.x; i < 16 * DIMC / 8; i += 256) dst[i] = src[i];
    }
    __syncthreads();

    const float *bias, *bn; float* dstp; int Cout, obase;
    if (orow0 < QKC)        { bias = bq; bn = bnq; dstp = qws; Cout = QKC; obase = 0;       }
    else if (orow0 < 2*QKC) { bias = bk; bn = bnk; dstp = kws; Cout = QKC; obase = QKC;     }
    else                    { bias = bv; bn = bnv; dstp = vws; Cout = DH_; obase = 2*QKC;   }

    const _Float16* arow = As + (lane & 15) * DIMC;
    const _Float16* bcol = xt16 + (size_t)col * DIMC;

    v8f acc = {};
    for (int k0 = 0; k0 < DIMC; k0 += 32) {
        __builtin_prefetch(bcol + k0 + 64, 0, 3);                // near prefetch
        const v16h a  = load_a_frag(arow, k0, lane);
        const v16h bf = load_b_frag(bcol, k0, lane);
        acc = __builtin_amdgcn_wmma_f32_16x16x32_f16(false, a, false, bf,
                                                     (short)0, acc, false, false);
    }
#pragma unroll
    for (int r = 0; r < 8; ++r) {
        const int o = orow0 - obase + r + ((lane & 16) ? 8 : 0);
        const float g  = bn[0 * Cout + o], be = bn[1 * Cout + o];
        const float mu = bn[2 * Cout + o], va = bn[3 * Cout + o];
        const float val = (acc[r] + bias[o] - mu) * rsqrtf(va + EPSB) * g + be;
        dstp[((size_t)b * Cout + o) * NTOK + n] = val;
        if (obase == 2 * QKC)                                    // v: f16 copy
            v16[((size_t)b * DH_ + o) * AMST + n] = (_Float16)val;
    }
}

// ---------------------------------------------------------------------------
// Kernel 2: 3x3 depthwise conv (SAME, cross-correlation) + BN -> ht16[(b,n)][ch]
// ---------------------------------------------------------------------------
__global__ __launch_bounds__(256)
void dwconv_bn(const float* __restrict__ vws, const float* __restrict__ wvl,
               const float* __restrict__ bvl, const float* __restrict__ bnvl,
               _Float16* __restrict__ ht16)
{
    const size_t idx = (size_t)blockIdx.x * 256 + threadIdx.x;   // B*N*DH
    const int ch = (int)(idx % DH_);
    const int n  = (int)((idx / DH_) % NTOK);
    const int b  = (int)(idx / ((size_t)DH_ * NTOK));
    const int y  = n / RES_, xx = n % RES_;
    const float* vimg = vws + ((size_t)b * DH_ + ch) * NTOK;
    const float* wf   = wvl + (size_t)ch * 9;
    float acc = bvl[ch];
#pragma unroll
    for (int dy = -1; dy <= 1; ++dy)
#pragma unroll
        for (int dx = -1; dx <= 1; ++dx) {
            const int yy = y + dy, xc = xx + dx;
            if (yy >= 0 && yy < RES_ && xc >= 0 && xc < RES_)
                acc += wf[(dy + 1) * 3 + (dx + 1)] * vimg[yy * RES_ + xc];
        }
    const float g  = bnvl[0*DH_+ch], be = bnvl[1*DH_+ch];
    const float mu = bnvl[2*DH_+ch], va = bnvl[3*DH_+ch];
    ht16[((size_t)b * NTOK + n) * DH_ + ch] =
        (_Float16)((acc - mu) * rsqrtf(va + EPSB) * g + be);
}

// ---------------------------------------------------------------------------
// Kernel 3: one block per (b, n): QK^T row for all 8 heads, +bias, th1 mix,
// wave32 softmax over m, th2 mix -> attn f16 (row stride 224, pad zeroed).
// ---------------------------------------------------------------------------
__global__ __launch_bounds__(256)
void attn_scores(const float* __restrict__ qws, const float* __restrict__ kws,
                 const float* __restrict__ ab, const int* __restrict__ bidx,
                 const float* __restrict__ th1w, const float* __restrict__ th1b,
                 const float* __restrict__ th2w, const float* __restrict__ th2b,
                 _Float16* __restrict__ attn)
{
    __shared__ float qrow[QKC];
    __shared__ float L[HEADS_][NTOK];
    __shared__ float S[HEADS_][NTOK];
    const int tid = threadIdx.x;
    const int b = blockIdx.x / NTOK;
    const int n = blockIdx.x % NTOK;

    qrow[tid] = qws[((size_t)b * QKC + tid) * NTOK + n];
    __syncthreads();

    for (int e = tid; e < HEADS_ * NTOK; e += 256) {
        const int h = e / NTOK, m = e % NTOK;
        const float* kcol = kws + ((size_t)b * QKC + h * KD) * NTOK + m;
        float dot = 0.f;
#pragma unroll
        for (int d = 0; d < KD; ++d) dot += qrow[h * KD + d] * kcol[(size_t)d * NTOK];
        L[h][m] = dot * SCALE_ + ab[h * NTOK + bidx[n * NTOK + m]];
    }
    __syncthreads();

    const int o = tid >> 5, lane = tid & 31;     // wave o -> output head o
    float t[7];
    float mx = -3.0e38f;
#pragma unroll
    for (int i = 0; i < 7; ++i) {
        const int m = lane + 32 * i;
        if (m < NTOK) {
            float s = th1b[o];
#pragma unroll
            for (int h = 0; h < HEADS_; ++h) s += th1w[o * HEADS_ + h] * L[h][m];
            t[i] = s; mx = fmaxf(mx, s);
        } else t[i] = -3.0e38f;
    }
    for (int d = 16; d > 0; d >>= 1) mx = fmaxf(mx, __shfl_xor(mx, d));
    float sum = 0.f;
#pragma unroll
    for (int i = 0; i < 7; ++i) {
        const int m = lane + 32 * i;
        if (m < NTOK) { t[i] = __expf(t[i] - mx); sum += t[i]; }
    }
    for (int d = 16; d > 0; d >>= 1) sum += __shfl_xor(sum, d);
    const float inv = 1.f / sum;
#pragma unroll
    for (int i = 0; i < 7; ++i) {
        const int m = lane + 32 * i;
        if (m < NTOK) S[o][m] = t[i] * inv;
    }
    __syncthreads();

    _Float16* arow = attn + ((size_t)(b * HEADS_ + o) * NTOK + n) * AMST;
#pragma unroll
    for (int i = 0; i < 7; ++i) {
        const int m = lane + 32 * i;                 // covers 0..223 (pad incl.)
        float a2 = 0.f;
        if (m < NTOK) {
            a2 = th2b[o];
#pragma unroll
            for (int h = 0; h < HEADS_; ++h) a2 += th2w[o * HEADS_ + h] * S[h][m];
        }
        arow[m] = (_Float16)a2;                      // pad written as exact 0
    }
}

// ---------------------------------------------------------------------------
// Kernel 4: attn @ V per head (WMMA, K=224 incl. zero pad). The shared 16-row
// attn A-tile (16 x 224 f16, contiguous rows) is DMA'd into LDS by the Tensor
// Data Mover from wave 0; the workgroup barrier publishes it. Epilogue fuses
// (+ v_local, ReLU) and stores f16 into ht16 = proj's B layout.
// grid = 64 * 8 * 13 = 6656
// ---------------------------------------------------------------------------
__global__ __launch_bounds__(256)
void attn_v(const _Float16* __restrict__ attn, const _Float16* __restrict__ v16,
            _Float16* __restrict__ ht16)
{
    __shared__ __align__(16) _Float16 As[16 * AMST];             // 7 KB
    const int lane = threadIdx.x & 31;
    const int wave = threadIdx.x >> 5;
    const int nt = blockIdx.x % 13;
    const int t  = blockIdx.x / 13;
    const int hd = t & 7;
    const int b  = t >> 3;
    const int n0 = nt * 16;
    const int e0 = wave * 16;

    if (threadIdx.x < 32) {     // one TDM transfer per workgroup (wave 0)
        tdm_load_tile_b16(As,
            attn + ((size_t)(b * HEADS_ + hd) * NTOK + n0) * AMST,
            /*tile_x=*/AMST, /*tile_y=*/16, /*row_stride=*/AMST);
    }
    __syncthreads();

    const _Float16* arow = As + (lane & 15) * AMST;
    const int ecol = e0 + (lane & 15);
    const _Float16* vcol = v16 + ((size_t)b * DH_ + hd * DV + ecol) * AMST;

    v8f acc = {};
    for (int k0 = 0; k0 < AMST; k0 += 32) {
        __builtin_prefetch(vcol + k0 + 64, 0, 3);
        const v16h a  = load_a_frag(arow, k0, lane);
        const v16h bf = load_b_frag(vcol, k0, lane);
        acc = __builtin_amdgcn_wmma_f32_16x16x32_f16(false, a, false, bf,
                                                     (short)0, acc, false, false);
    }
#pragma unroll
    for (int r = 0; r < 8; ++r) {
        const int n = n0 + r + ((lane & 16) ? 8 : 0);
        if (n < NTOK) {
            const int ch = hd * DV + ecol;
            const size_t idx = ((size_t)b * NTOK + n) * DH_ + ch;
            const float o = acc[r] + (float)ht16[idx];           // + v_local
            ht16[idx] = (_Float16)(o > 0.f ? o : 0.f);           // fused ReLU
        }
    }
}

// ---------------------------------------------------------------------------
// Kernel 5: d_out = BN(wp @ relu(h) + bp).  384x1024x12544 WMMA GEMM; A-tile
// (16x1024 f16 = 32 KB) staged in LDS, shared by the block's 8 waves.
// grid = 24 * 98 = 2352
// ---------------------------------------------------------------------------
__global__ __launch_bounds__(256)
void proj_gemm(const _Float16* __restrict__ ht16, const _Float16* __restrict__ wp16,
               const float* __restrict__ bp, const float* __restrict__ bnp,
               float* __restrict__ out)
{
    __shared__ __align__(16) _Float16 As[16 * DH_];              // 32 KB
    const int lane = threadIdx.x & 31;
    const int wave = threadIdx.x >> 5;
    const int ot = blockIdx.x % 24;
    const int jt = (blockIdx.x / 24) * 8 + wave;
    const int col = jt * 16 + (lane & 15);                       // = b*196+n
    const int b = col / NTOK, n = col % NTOK;

    {   // cooperative A-tile copy: 16 rows x 1024 halves, contiguous
        const uint4* src = (const uint4*)(wp16 + (size_t)ot * 16 * DH_);
        uint4* dst = (uint4*)As;
        for (int i = threadIdx.x; i < 16 * DH_ / 8; i += 256) dst[i] = src[i];
    }
    __syncthreads();

    const _Float16* arow = As + (lane & 15) * DH_;
    const _Float16* bcol = ht16 + (size_t)col * DH_;

    v8f acc = {};
    for (int k0 = 0; k0 < DH_; k0 += 32) {
        __builtin_prefetch(bcol + k0 + 64, 0, 3);
        const v16h a  = load_a_frag(arow, k0, lane);
        const v16h bf = load_b_frag(bcol, k0, lane);
        acc = __builtin_amdgcn_wmma_f32_16x16x32_f16(false, a, false, bf,
                                                     (short)0, acc, false, false);
    }
#pragma unroll
    for (int r = 0; r < 8; ++r) {
        const int o = ot * 16 + r + ((lane & 16) ? 8 : 0);
        const float g  = bnp[0*DIMC+o], be = bnp[1*DIMC+o];
        const float mu = bnp[2*DIMC+o], va = bnp[3*DIMC+o];
        const float val = (acc[r] + bp[o] - mu) * rsqrtf(va + EPSB) * g + be;
        out[((size_t)b * DIMC + o) * NTOK + n] = val;
    }
}

// ---------------------------------------------------------------------------
extern "C" void kernel_launch(void* const* d_in, const int* in_sizes, int n_in,
                              void* d_out, int out_size, void* d_ws, size_t ws_size,
                              hipStream_t stream)
{
    const float* x    = (const float*)d_in[0];
    const float* wq   = (const float*)d_in[1];
    const float* bq   = (const float*)d_in[2];
    const float* bnq  = (const float*)d_in[3];
    const float* wk   = (const float*)d_in[4];
    const float* bk   = (const float*)d_in[5];
    const float* bnk  = (const float*)d_in[6];
    const float* wv   = (const float*)d_in[7];
    const float* bv   = (const float*)d_in[8];
    const float* bnv  = (const float*)d_in[9];
    const float* wvl  = (const float*)d_in[10];
    const float* bvl  = (const float*)d_in[11];
    const float* bnvl = (const float*)d_in[12];
    const float* th1w = (const float*)d_in[13];
    const float* th1b = (const float*)d_in[14];
    const float* th2w = (const float*)d_in[15];
    const float* th2b = (const float*)d_in[16];
    const float* wp   = (const float*)d_in[17];
    const float* bp   = (const float*)d_in[18];
    const float* bnp  = (const float*)d_in[19];
    const float* ab   = (const float*)d_in[20];
    const int*   bidx = (const int*)d_in[21];
    float* out = (float*)d_out;

    constexpr size_t QKE  = (size_t)B_ * QKC * NTOK;                 // 3,211,264 f32
    constexpr size_t VE   = (size_t)B_ * DH_ * NTOK;                 // 12,845,056
    constexpr size_t AE   = (size_t)B_ * HEADS_ * NTOK * AMST + 16 * AMST; // f16
    constexpr size_t V16E = (size_t)B_ * DH_ * AMST;                 // f16
    constexpr size_t HTE  = VE;                                      // f16
    constexpr size_t W16E = (size_t)1536 * DIMC;                     // f16
    constexpr size_t WP16E= (size_t)DIMC * DH_;                      // f16
    constexpr size_t XTE  = (size_t)B_ * NTOK * DIMC;                // f16
    constexpr size_t NEED = (2 * QKE + VE) * sizeof(float)
                          + (AE + V16E + HTE + W16E + WP16E + XTE) * sizeof(_Float16);
    if (ws_size < NEED) return;                                      // ~180 MB

    float* qws = (float*)d_ws;
    float* kws = qws + QKE;
    float* vws = kws + QKE;
    _Float16* attn = (_Float16*)(vws + VE);
    _Float16* v16  = attn + AE;
    _Float16* ht16 = v16 + V16E;
    _Float16* w16  = ht16 + HTE;
    _Float16* wp16 = w16 + W16E;
    _Float16* xt16 = wp16 + WP16E;

    pack_weights<<< 3840, 256, 0, stream>>>(wq, wk, wv, wp, w16, wp16);
    pack_x      <<<18816, 256, 0, stream>>>(x, xt16);
    qkv_gemm    <<< 9408, 256, 0, stream>>>(w16, xt16, bq, bnq, bk, bnk, bv, bnv,
                                            qws, kws, vws, v16);
    dwconv_bn   <<<50176, 256, 0, stream>>>(vws, wvl, bvl, bnvl, ht16);
    attn_scores <<<12544, 256, 0, stream>>>(qws, kws, ab, bidx,
                                            th1w, th1b, th2w, th2b, attn);
    attn_v      <<< 6656, 256, 0, stream>>>(attn, v16, ht16);
    proj_gemm   <<< 2352, 256, 0, stream>>>(ht16, wp16, bp, bnp, out);

    (void)in_sizes; (void)n_in; (void)out_size;
}
